// Recurrence_2860448219542
// MI455X (gfx1250) — compile-verified, matrix operand order checked by reading, not run on hardware
//
#include <hip/hip_runtime.h>
#include <math.h>

// ---------------------------------------------------------------------------
// GRU rollout + actor/critic MLP for MI455X (gfx1250, wave32, WMMA bf16)
//  - x-side gates hoisted out of the scan into one batched WMMA GEMM
//  - 32x64 wave tiles, double-buffered fragment loads for load/MMA overlap
// ---------------------------------------------------------------------------

typedef __bf16 bf16;
typedef __attribute__((ext_vector_type(16))) __bf16 v16bf;
typedef __attribute__((ext_vector_type(8)))  __bf16 v8bf;
typedef __attribute__((ext_vector_type(8)))  float  v8f;

#define T_    128
#define N_    256
#define H_    512
#define DIN_  255
#define G3_   1536          // 3*H
#define KX_   256           // DIN padded to multiple of 32
#define PACK_ 517           // H + 5
#define TN_   (T_ * N_)     // 32768

// ---- bf16 <-> f32 bit helpers (round-to-nearest-even on pack) -------------
__device__ __forceinline__ bf16 f2bf(float f) {
  unsigned u = __float_as_uint(f);
  u += 0x7FFFu + ((u >> 16) & 1u);
  union { unsigned short s; bf16 b; } c;
  c.s = (unsigned short)(u >> 16);
  return c.b;
}
__device__ __forceinline__ float bf2f(bf16 b) {
  union { unsigned short s; bf16 b; } c; c.b = b;
  return __uint_as_float(((unsigned)c.s) << 16);
}

// ---- WMMA 16x16x32 bf16 fragment load -------------------------------------
// 16-bit A (and B, mirrored with N on lanes) layout, ISA 7.12.2:
//   lanes 0-15 : row = lane,    K = k0+{0..7} and k0+{16..23}
//   lanes 16-31: row = lane-16, K = k0+{8..15} and k0+{24..31}
__device__ __forceinline__ v16bf load_frag(const bf16* __restrict__ base,
                                           int ld, int row0, int k0, int lane) {
  const int half = lane >> 4, r = lane & 15;
  const bf16* p = base + (long)(row0 + r) * ld + k0 + half * 8;
  v8bf lo = *(const v8bf*)(p);        // global_load_b128
  v8bf hi = *(const v8bf*)(p + 16);   // global_load_b128
  v16bf f;
#pragma unroll
  for (int i = 0; i < 8; ++i) { f[i] = lo[i]; f[i + 8] = hi[i]; }
  return f;
}

// Load one double-buffer stage: 2 A-fragments (32 M rows) + 4 B-fragments (64 N cols)
__device__ __forceinline__ void load_set(
    const bf16* __restrict__ A, int lda, int row0,
    const bf16* __restrict__ W, int ldw, int col0,
    int k0, int lane, v16bf a[2], v16bf b[4]) {
#pragma unroll
  for (int m = 0; m < 2; ++m) a[m] = load_frag(A, lda, row0 + (m << 4), k0, lane);
#pragma unroll
  for (int s = 0; s < 4; ++s) b[s] = load_frag(W, ldw, col0 + (s << 4), k0, lane);
}

__device__ __forceinline__ void mma_set(const v16bf a[2], const v16bf b[4],
                                        v8f acc[2][4]) {
#pragma unroll
  for (int m = 0; m < 2; ++m)
#pragma unroll
    for (int s = 0; s < 4; ++s)
      acc[m][s] = __builtin_amdgcn_wmma_f32_16x16x32_bf16(
          false, a[m], false, b[s], (short)0, acc[m][s], false, false);
}

// ---- Generic GEMM: C[M x Nout] = act(A[M x K] @ W[Nout x K]^T + bias) -----
// MODE 0: f32 out, no act.  MODE 1: bf16 out, tanh.  MODE 2: bf16 out, no act.
// Wave tile: 32(M) x 64(N); K stepped by 64 with two pipelined 32-wide stages.
// Requires M % 32 == 0, Nout % 64 == 0, K % 64 == 0.
template <int MODE>
__global__ __launch_bounds__(256) void gemm_wmma(
    const bf16* __restrict__ A, int lda,
    const bf16* __restrict__ W, int ldw,
    const float* __restrict__ bias,
    float* __restrict__ Cf, bf16* __restrict__ Cb, int ldc,
    int M, int Nout, int K) {
  const int lane  = threadIdx.x & 31;
  const int wave  = blockIdx.x * (blockDim.x >> 5) + (threadIdx.x >> 5);
  const int tilesN = Nout >> 6;
  const int nTiles = (M >> 5) * tilesN;
  if (wave >= nTiles) return;
  const int row0 = (wave / tilesN) << 5;
  const int col0 = (wave % tilesN) << 6;

  v8f acc[2][4] = {};
  v16bf aP[2], bP[4], aQ[2], bQ[4];

  load_set(A, lda, row0, W, ldw, col0, 0, lane, aP, bP);
  for (int k0 = 0; k0 + 64 <= K; k0 += 64) {
    load_set(A, lda, row0, W, ldw, col0, k0 + 32, lane, aQ, bQ);  // prefetch
    mma_set(aP, bP, acc);
    if (k0 + 64 < K)
      load_set(A, lda, row0, W, ldw, col0, k0 + 64, lane, aP, bP); // prefetch
    mma_set(aQ, bQ, acc);
  }

  // C/D layout: acc[m][s][i] of lane L -> row = row0+16m+i+8*(L>>4),
  //                                       col = col0+16s+(L&15)
  const int half = lane >> 4, cn = lane & 15;
#pragma unroll
  for (int m = 0; m < 2; ++m) {
#pragma unroll
    for (int s = 0; s < 4; ++s) {
      const int col = col0 + (s << 4) + cn;
      const float bv = bias ? bias[col] : 0.f;
#pragma unroll
      for (int i = 0; i < 8; ++i) {
        const int row = row0 + (m << 4) + (half << 3) + i;
        const float v = acc[m][s][i] + bv;
        if (MODE == 0)      Cf[(long)row * ldc + col] = v;
        else if (MODE == 1) Cb[(long)row * ldc + col] = f2bf(tanhf(v));
        else                Cb[(long)row * ldc + col] = f2bf(v);
      }
    }
  }
}

// ---- GRU gate fusion: r,z,n gates + hidden update -------------------------
// Gxb: precomputed x-side gates for this step (bf16), Gh: h-side gates (f32).
__global__ __launch_bounds__(256) void gru_gate(
    const bf16* __restrict__ Gxb, const float* __restrict__ Gh,
    float* __restrict__ Hf, bf16* __restrict__ Hb,
    bf16* __restrict__ Hsb_t, float* __restrict__ out_t) {
  const int i = blockIdx.x * blockDim.x + threadIdx.x;
  if (i >= N_ * H_) return;
  const int n = i >> 9, j = i & (H_ - 1);
  const bf16*  gx = Gxb + (long)n * G3_;
  const float* gh = Gh  + (long)n * G3_;
  const float r  = 1.f / (1.f + expf(-(bf2f(gx[j]) + gh[j])));
  const float z  = 1.f / (1.f + expf(-(bf2f(gx[H_ + j]) + gh[H_ + j])));
  const float nn = tanhf(bf2f(gx[2 * H_ + j]) + r * gh[2 * H_ + j]);
  const float h  = (1.f - z) * nn + z * Hf[i];
  Hf[i] = h;
  Hb[i] = f2bf(h);
  Hsb_t[i] = f2bf(h);
  out_t[(long)n * PACK_ + 4 + j] = h;   // hs slot of packed output
}

// ---- GEMV heads: out[row] = X[row,:] . w + b (K = 512), one wave per row --
__global__ __launch_bounds__(256) void gemv512(
    const bf16* __restrict__ X, const float* __restrict__ w,
    const float* __restrict__ b, float* __restrict__ out, int M) {
  const int lane = threadIdx.x & 31;
  const int row  = blockIdx.x * (blockDim.x >> 5) + (threadIdx.x >> 5);
  if (row >= M) return;
  const bf16* x = X + (long)row * H_;
  float s = 0.f;
#pragma unroll
  for (int hlf = 0; hlf < 2; ++hlf) {
    const int j = (lane << 3) + hlf * 256;
    v8bf xv = *(const v8bf*)(x + j);      // 16B load
#pragma unroll
    for (int i = 0; i < 8; ++i) s += bf2f(xv[i]) * w[j + i];
  }
#pragma unroll
  for (int off = 16; off; off >>= 1) s += __shfl_xor(s, off, 32);
  if (lane == 0) out[row] = s + b[0];
}

// ---- Repack helpers -------------------------------------------------------
__global__ __launch_bounds__(256) void pack_pad(
    const float* __restrict__ src, int srcStride, int srcCols,
    bf16* __restrict__ dst, int dstCols, long rows) {
  const long i = (long)blockIdx.x * blockDim.x + threadIdx.x;
  if (i >= rows * dstCols) return;
  const long r = i / dstCols;
  const int  c = (int)(i - r * dstCols);
  const float v = (c < srcCols) ? src[r * srcStride + c] : 0.f;
  dst[i] = f2bf(v);
}

__global__ __launch_bounds__(256) void cvt_bf16(
    const float* __restrict__ src, bf16* __restrict__ dst, long n) {
  const long i = (long)blockIdx.x * blockDim.x + threadIdx.x;
  if (i < n) dst[i] = f2bf(src[i]);
}

__global__ __launch_bounds__(256) void init_h(
    const float* __restrict__ hx, float* __restrict__ Hf, bf16* __restrict__ Hb) {
  const int i = blockIdx.x * blockDim.x + threadIdx.x;
  if (i >= N_ * H_) return;
  const int n = i >> 9, j = i & (H_ - 1);
  const float v = hx[(long)n * PACK_ + 4 + j];
  Hf[i] = v; Hb[i] = f2bf(v);
}

// ---- Final assembly: a, loc, scale, v, p slots of packed output -----------
__global__ __launch_bounds__(256) void assemble(
    const float* __restrict__ inputs, const float* __restrict__ eps,
    const float* __restrict__ hx, const float* __restrict__ logstd,
    const float* __restrict__ locbuf, const float* __restrict__ vbuf,
    float* __restrict__ out) {
  const int i = blockIdx.x * blockDim.x + threadIdx.x;
  if (i >= TN_) return;
  const int n = i & (N_ - 1);
  const float scale  = expf(logstd[0]);
  const float loc    = locbuf[i];
  const float sample = loc + scale * eps[i];
  const float act    = inputs[(long)i * (DIN_ + 1) + DIN_];
  const float a      = (act < 0.f) ? sample : act;
  float* o = out + (long)i * PACK_;
  o[0] = a;
  o[1] = loc;
  o[2] = scale;
  o[3] = vbuf[i];
  o[PACK_ - 1] = hx[(long)n * PACK_ + (PACK_ - 1)];   // p broadcast
}

// second tuple element = packed[-1:] (copy last time slice)
__global__ __launch_bounds__(256) void tail_copy(float* __restrict__ out) {
  const int i = blockIdx.x * blockDim.x + threadIdx.x;
  if (i >= N_ * PACK_) return;
  out[(long)T_ * N_ * PACK_ + i] = out[(long)(T_ - 1) * N_ * PACK_ + i];
}

// ---------------------------------------------------------------------------
extern "C" void kernel_launch(void* const* d_in, const int* in_sizes, int n_in,
                              void* d_out, int out_size, void* d_ws, size_t ws_size,
                              hipStream_t stream) {
  const float* inputs = (const float*)d_in[0];   // (T,N,256)
  const float* hx     = (const float*)d_in[1];   // (1,N,517)
  const float* eps    = (const float*)d_in[2];   // (T,N,1)
  const float* Wih    = (const float*)d_in[3];   // (1536,255)
  const float* Whh    = (const float*)d_in[4];   // (1536,512)
  const float* bih    = (const float*)d_in[5];
  const float* bhh    = (const float*)d_in[6];
  const float* Wa0    = (const float*)d_in[7];
  const float* ba0    = (const float*)d_in[8];
  const float* Wa1    = (const float*)d_in[9];
  const float* ba1    = (const float*)d_in[10];
  const float* Wloc   = (const float*)d_in[11];
  const float* bloc   = (const float*)d_in[12];
  const float* logstd = (const float*)d_in[13];
  const float* Wc0    = (const float*)d_in[14];
  const float* bc0    = (const float*)d_in[15];
  const float* Wc1    = (const float*)d_in[16];
  const float* bc1    = (const float*)d_in[17];
  const float* Wv     = (const float*)d_in[18];
  const float* bv     = (const float*)d_in[19];
  float* out = (float*)d_out;

  // Workspace carve-out (256B-aligned slabs)
  char* wp = (char*)d_ws;
  auto alloc = [&](size_t bytes) -> void* {
    void* r = (void*)wp;
    wp += (bytes + 255) & ~(size_t)255;
    return r;
  };
  bf16*  Xb     = (bf16*) alloc((size_t)TN_ * KX_ * 2);   // obs bf16, K padded
  bf16*  Wihb   = (bf16*) alloc((size_t)G3_ * KX_ * 2);
  bf16*  Whhb   = (bf16*) alloc((size_t)G3_ * H_ * 2);
  bf16*  Wa0b   = (bf16*) alloc((size_t)H_ * H_ * 2);
  bf16*  Wa1b   = (bf16*) alloc((size_t)H_ * H_ * 2);
  bf16*  Wc0b   = (bf16*) alloc((size_t)H_ * H_ * 2);
  bf16*  Wc1b   = (bf16*) alloc((size_t)H_ * H_ * 2);
  bf16*  Hb     = (bf16*) alloc((size_t)N_ * H_ * 2);     // recurrent state bf16
  float* Hf     = (float*)alloc((size_t)N_ * H_ * 4);     // recurrent state f32
  float* Gh     = (float*)alloc((size_t)N_ * G3_ * 4);    // h-side gates (per step)
  bf16*  Hsb    = (bf16*) alloc((size_t)TN_ * H_ * 2);    // all hidden states
  float* locbuf = (float*)alloc((size_t)TN_ * 4);
  float* vbuf   = (float*)alloc((size_t)TN_ * 4);
  // Big slab: x-side gates for ALL steps (bf16, ~100 MB), consumed by the
  // scan; the MLP scratch buffers alias into it afterwards.
  bf16*  Gxall  = (bf16*) alloc((size_t)TN_ * G3_ * 2);
  bf16*  T0b    = Gxall;                                  // alias (post-scan)
  bf16*  T1b    = Gxall + (size_t)TN_ * H_;               // alias (post-scan)
  (void)ws_size; (void)in_sizes; (void)n_in; (void)out_size;

  auto cdiv = [](long a, long b) { return (int)((a + b - 1) / b); };

  // ---- 1. One-time bf16 repacks ------------------------------------------
  pack_pad<<<cdiv((long)TN_ * KX_, 256), 256, 0, stream>>>(
      inputs, DIN_ + 1, DIN_, Xb, KX_, TN_);
  pack_pad<<<cdiv((long)G3_ * KX_, 256), 256, 0, stream>>>(
      Wih, DIN_, DIN_, Wihb, KX_, G3_);
  cvt_bf16<<<cdiv((long)G3_ * H_, 256), 256, 0, stream>>>(Whh, Whhb, (long)G3_ * H_);
  cvt_bf16<<<cdiv((long)H_ * H_, 256), 256, 0, stream>>>(Wa0, Wa0b, (long)H_ * H_);
  cvt_bf16<<<cdiv((long)H_ * H_, 256), 256, 0, stream>>>(Wa1, Wa1b, (long)H_ * H_);
  cvt_bf16<<<cdiv((long)H_ * H_, 256), 256, 0, stream>>>(Wc0, Wc0b, (long)H_ * H_);
  cvt_bf16<<<cdiv((long)H_ * H_, 256), 256, 0, stream>>>(Wc1, Wc1b, (long)H_ * H_);
  init_h<<<cdiv(N_ * H_, 256), 256, 0, stream>>>(hx, Hf, Hb);

  // ---- 2. Hoisted x-side gates: one batched GEMM over all T*N rows --------
  // Gxall = inputs @ Wih^T + bih   (M=32768, Nout=1536, K=256; ~26 GFLOP)
  {
    const int tiles  = (TN_ / 32) * (G3_ / 64);
    gemm_wmma<2><<<cdiv(tiles, 8), 256, 0, stream>>>(
        Xb, KX_, Wihb, KX_, bih, nullptr, Gxall, G3_, TN_, G3_, KX_);
  }

  // ---- 3. Sequential GRU scan (only the recurrent GEMM remains) -----------
  const int gruTiles  = (N_ / 32) * (G3_ / 64);           // 192 wave tiles
  const int gruBlocks = cdiv(gruTiles, 8);                // 24 blocks
  for (int t = 0; t < T_; ++t) {
    gemm_wmma<0><<<gruBlocks, 256, 0, stream>>>(
        Hb, H_, Whhb, H_, bhh, Gh, nullptr, G3_, N_, G3_, H_);
    gru_gate<<<cdiv(N_ * H_, 256), 256, 0, stream>>>(
        Gxall + (long)t * N_ * G3_, Gh, Hf, Hb,
        Hsb + (long)t * N_ * H_, out + (long)t * N_ * PACK_);
  }

  // ---- 4. Batched MLP heads over all T*N rows (compute-bound WMMA) --------
  const int mlpTiles  = (TN_ / 32) * (H_ / 64);
  const int mlpBlocks = cdiv(mlpTiles, 8);
  // actor
  gemm_wmma<1><<<mlpBlocks, 256, 0, stream>>>(
      Hsb, H_, Wa0b, H_, ba0, nullptr, T0b, H_, TN_, H_, H_);
  gemm_wmma<1><<<mlpBlocks, 256, 0, stream>>>(
      T0b, H_, Wa1b, H_, ba1, nullptr, T1b, H_, TN_, H_, H_);
  gemv512<<<cdiv(TN_, 8), 256, 0, stream>>>(T1b, Wloc, bloc, locbuf, TN_);
  // critic (reuse scratch)
  gemm_wmma<1><<<mlpBlocks, 256, 0, stream>>>(
      Hsb, H_, Wc0b, H_, bc0, nullptr, T0b, H_, TN_, H_, H_);
  gemm_wmma<1><<<mlpBlocks, 256, 0, stream>>>(
      T0b, H_, Wc1b, H_, bc1, nullptr, T1b, H_, TN_, H_, H_);
  gemv512<<<cdiv(TN_, 8), 256, 0, stream>>>(T1b, Wv, bv, vbuf, TN_);

  // ---- 5. Assemble packed output + tail slice -----------------------------
  assemble<<<cdiv(TN_, 256), 256, 0, stream>>>(
      inputs, eps, hx, logstd, locbuf, vbuf, out);
  tail_copy<<<cdiv(N_ * PACK_, 256), 256, 0, stream>>>(out);
}
